// MoE_73220602462487
// MI455X (gfx1250) — matrix operand order repaired and checked by reading, not compiled, not run on hardware
//
#include <hip/hip_runtime.h>
#include <hip/hip_bf16.h>
#include <math.h>

// ---------------------------------------------------------------------------
// MoE (top-2 of 8 experts + shared expert) for gfx1250 (MI455X).
// All heavy math via v_wmma_f32_16x16x32_bf16 (wave32 WMMA).
// N=2048 tokens, D=2048 model dim, H=1024 hidden, E=8 experts, K=2.
// Block tile 64(M) x 128(N), 8 waves as 2(M) x 4(N); per-wave tile 32x32
// (2 A-frags x 2 B-cols) so each B fragment feeds 2 WMMAs.
// ---------------------------------------------------------------------------

#define NTOK 2048
#define DIM  2048
#define HID  1024
#define NEXP 8

typedef __attribute__((ext_vector_type(16))) __bf16 bf16x16;
typedef __attribute__((ext_vector_type(8)))  __bf16 bf16x8;
typedef __attribute__((ext_vector_type(8)))  float  f32x8;

__device__ __forceinline__ __bf16 f2bf(float f) {
  unsigned u = __builtin_bit_cast(unsigned, f);
  unsigned r = u + 0x7FFFu + ((u >> 16) & 1u);   // round-to-nearest-even
  unsigned short h = (unsigned short)(r >> 16);
  return __builtin_bit_cast(__bf16, h);
}

__device__ __forceinline__ f32x8 wmma_bf16f32(bf16x16 a, bf16x16 b, f32x8 c) {
  // (neg_a, A, neg_b, B, c_mod, C, reuse_a, reuse_b)
  return __builtin_amdgcn_wmma_f32_16x16x32_bf16(false, a, false, b,
                                                 (short)0, c, false, false);
}

// A-matrix 16x32 bf16 fragment (ISA 7.12.2): lanes 0-15 hold K=[0..7],[16..23],
// lanes 16-31 hold K=[8..15],[24..31] of row (lane&15).
__device__ __forceinline__ bf16x16 frag_a(const __bf16* __restrict__ row,
                                          int k0, int hsel) {
  bf16x8 lo = *(const bf16x8*)(row + k0 + hsel * 8);
  bf16x8 hi = *(const bf16x8*)(row + k0 + 16 + hsel * 8);
  return __builtin_shufflevector(lo, hi, 0,1,2,3,4,5,6,7,8,9,10,11,12,13,14,15);
}

// B-matrix 32x16 bf16 fragment: lane holds column n=(lane&15), K=[0..7],[16..23]
// for lanes 0-15 and K=[8..15],[24..31] for lanes 16-31 (same as A). A row-major weight row W[n][k..]
// is exactly one B column, so we load it directly from global.
__device__ __forceinline__ bf16x16 frag_b(const __bf16* __restrict__ col,
                                          int k0, int hsel) {
  bf16x8 lo = *(const bf16x8*)(col + k0 + hsel * 8);
  bf16x8 hi = *(const bf16x8*)(col + k0 + 16 + hsel * 8);
  return __builtin_shufflevector(lo, hi, 0,1,2,3,4,5,6,7,8,9,10,11,12,13,14,15);
}

// ------------------------------- cast kernel -------------------------------
__global__ void cast_f32_bf16(const float* __restrict__ in,
                              __bf16* __restrict__ out, long n) {
  long i = (long)blockIdx.x * blockDim.x + threadIdx.x;
  long stride = (long)gridDim.x * blockDim.x;
  for (; i < n; i += stride) out[i] = f2bf(in[i]);
}

__global__ void zero_counts(int* cnt) {
  if (threadIdx.x < NEXP) cnt[threadIdx.x] = 0;
}

// -------------------------------- router -----------------------------------
// One wave per token: logits = x . gate (E=8), sigmoid, top-2 on score+bias,
// atomic compaction into per-expert lists.
__global__ __launch_bounds__(256)
void router_kernel(const float* __restrict__ x, const float* __restrict__ gate,
                   const float* __restrict__ ebias, int* __restrict__ cnt,
                   int* __restrict__ tok_list, float* __restrict__ score_list,
                   int* __restrict__ dst_list) {
  int t = (int)((blockIdx.x * blockDim.x + threadIdx.x) >> 5);
  int lane = threadIdx.x & 31;
  if (t >= NTOK) return;

  float acc[NEXP];
#pragma unroll
  for (int e = 0; e < NEXP; ++e) acc[e] = 0.f;

  const float* xr = x + (size_t)t * DIM;
  for (int d = lane; d < DIM; d += 32) {
    float xv = xr[d];
    const float* g = gate + (size_t)d * NEXP;
#pragma unroll
    for (int e = 0; e < NEXP; ++e) acc[e] += xv * g[e];
  }
#pragma unroll
  for (int off = 16; off > 0; off >>= 1) {
#pragma unroll
    for (int e = 0; e < NEXP; ++e) acc[e] += __shfl_xor(acc[e], off, 32);
  }
  if (lane == 0) {
    float sc[NEXP], bsc[NEXP];
#pragma unroll
    for (int e = 0; e < NEXP; ++e) {
      sc[e] = 1.f / (1.f + expf(-acc[e]));
      bsc[e] = sc[e] + ebias[e];
    }
    int b0 = 0; float v0 = bsc[0];
#pragma unroll
    for (int e = 1; e < NEXP; ++e) if (bsc[e] > v0) { v0 = bsc[e]; b0 = e; }
    int b1 = -1; float v1 = -3.4e38f;
#pragma unroll
    for (int e = 0; e < NEXP; ++e)
      if (e != b0 && bsc[e] > v1) { v1 = bsc[e]; b1 = e; }
    int pick[2] = { b0, b1 };
#pragma unroll
    for (int k = 0; k < 2; ++k) {
      int e = pick[k];
      int i = atomicAdd(&cnt[e], 1);
      tok_list[e * NTOK + i] = t;
      score_list[e * NTOK + i] = sc[e];
      dst_list[e * NTOK + i] = t * 2 + k;
    }
  }
}

// --------------------------- GEMM1: x @ {w1,w3}^T + SwiGLU -----------------
// Block tile: 64 tokens x 128 hidden; wave (waveM,waveN) covers 32x32 for
// BOTH gate (w1) and up (w3) paths -> 8 accumulator fragments (64 VGPRs).
template <bool ROUTED>
__global__ __launch_bounds__(256)
void ffn_gemm1(const __bf16* __restrict__ xb,
               const __bf16* __restrict__ w1b, const __bf16* __restrict__ w3b,
               const int* __restrict__ cnt, const int* __restrict__ tok_list,
               const float* __restrict__ score_list,
               const int* __restrict__ dst_list,
               __bf16* __restrict__ h_out) {
  const int e = blockIdx.z;
  const int count = ROUTED ? cnt[e] : NTOK;
  const int m0 = blockIdx.x * 64;
  if (m0 >= count) return;
  const int n0 = blockIdx.y * 128;

  const int tid = threadIdx.x;
  const int lane = tid & 31;
  const int wid = tid >> 5;
  const int waveM = wid >> 2;       // 0..1 -> 32 rows each
  const int waveN = wid & 3;        // 0..3 -> 32 cols each
  const int hsel = lane >> 4;
  const int lm = lane & 15;

  __shared__ float s_s[64];
  __shared__ int s_tok[64];
  __shared__ int s_dst[64];
  if (tid < 64) {
    int gm = m0 + tid;
    if (ROUTED) {
      bool v = gm < count;
      int ai = e * NTOK + (v ? gm : 0);
      s_tok[tid] = v ? tok_list[ai] : 0;
      s_s[tid]   = v ? score_list[ai] : 0.f;
      s_dst[tid] = v ? dst_list[ai] : 0;
    } else {
      s_tok[tid] = gm; s_s[tid] = 1.f; s_dst[tid] = gm;
    }
  }
  __syncthreads();

  const __bf16* w1e = w1b + (size_t)e * HID * DIM;
  const __bf16* w3e = w3b + (size_t)e * HID * DIM;

  const __bf16* aptr[2];
#pragma unroll
  for (int mi = 0; mi < 2; ++mi)
    aptr[mi] = xb + (size_t)s_tok[waveM * 32 + mi * 16 + lm] * DIM;

  const __bf16* b1p[2];
  const __bf16* b3p[2];
#pragma unroll
  for (int f = 0; f < 2; ++f) {
    int wrow = n0 + waveN * 32 + f * 16 + lm;
    b1p[f] = w1e + (size_t)wrow * DIM;
    b3p[f] = w3e + (size_t)wrow * DIM;
  }

  f32x8 cg[2][2], cu[2][2];
#pragma unroll
  for (int mi = 0; mi < 2; ++mi)
#pragma unroll
    for (int f = 0; f < 2; ++f) { cg[mi][f] = (f32x8)(0.f); cu[mi][f] = (f32x8)(0.f); }

  for (int k0 = 0; k0 < DIM; k0 += 32) {
    bf16x16 a0 = frag_a(aptr[0], k0, hsel);
    bf16x16 a1 = frag_a(aptr[1], k0, hsel);
#pragma unroll
    for (int f = 0; f < 2; ++f) {
      bf16x16 b1 = frag_b(b1p[f], k0, hsel);
      cg[0][f] = wmma_bf16f32(a0, b1, cg[0][f]);
      cg[1][f] = wmma_bf16f32(a1, b1, cg[1][f]);
      bf16x16 b3 = frag_b(b3p[f], k0, hsel);
      cu[0][f] = wmma_bf16f32(a0, b3, cu[0][f]);
      cu[1][f] = wmma_bf16f32(a1, b3, cu[1][f]);
    }
  }

  // Epilogue: h = silu(s*g) * (s*u); C/D layout: VGPR r holds row r + 8*hsel.
#pragma unroll
  for (int mi = 0; mi < 2; ++mi) {
#pragma unroll
    for (int r = 0; r < 8; ++r) {
      int ml = waveM * 32 + mi * 16 + r + 8 * hsel;
      if (ROUTED && (m0 + ml) >= count) continue;
      float s = s_s[ml];
      size_t drow = (size_t)s_dst[ml] * HID;
#pragma unroll
      for (int f = 0; f < 2; ++f) {
        float sg = s * cg[mi][f][r];
        float hv = sg * (1.f / (1.f + expf(-sg))) * (s * cu[mi][f][r]);
        h_out[drow + n0 + waveN * 32 + f * 16 + lm] = f2bf(hv);
      }
    }
  }
}

// --------------------------- GEMM2: h @ w2^T -> out ------------------------
// Shared path plain-stores (initializes d_out); routed path atomicAdds.
template <bool ROUTED>
__global__ __launch_bounds__(256)
void ffn_gemm2(const __bf16* __restrict__ hbuf, const __bf16* __restrict__ w2b,
               const int* __restrict__ cnt, const int* __restrict__ tok_list,
               const int* __restrict__ dst_list, float* __restrict__ out) {
  const int e = blockIdx.z;
  const int count = ROUTED ? cnt[e] : NTOK;
  const int m0 = blockIdx.x * 64;
  if (m0 >= count) return;
  const int n0 = blockIdx.y * 128;   // over DIM

  const int tid = threadIdx.x;
  const int lane = tid & 31;
  const int wid = tid >> 5;
  const int waveM = wid >> 2;
  const int waveN = wid & 3;
  const int hsel = lane >> 4;
  const int lm = lane & 15;

  __shared__ int s_tok[64];   // output token row
  __shared__ int s_hrow[64];  // A row index in hbuf
  if (tid < 64) {
    int gm = m0 + tid;
    if (ROUTED) {
      bool v = gm < count;
      int ai = e * NTOK + (v ? gm : 0);
      s_tok[tid]  = v ? tok_list[ai] : 0;
      s_hrow[tid] = v ? dst_list[ai] : 0;
    } else {
      s_tok[tid] = gm; s_hrow[tid] = gm;
    }
  }
  __syncthreads();

  const __bf16* w2e = w2b + (size_t)e * DIM * HID;

  const __bf16* aptr[2];
#pragma unroll
  for (int mi = 0; mi < 2; ++mi)
    aptr[mi] = hbuf + (size_t)s_hrow[waveM * 32 + mi * 16 + lm] * HID;

  const __bf16* bp[2];
#pragma unroll
  for (int f = 0; f < 2; ++f) {
    int wrow = n0 + waveN * 32 + f * 16 + lm;
    bp[f] = w2e + (size_t)wrow * HID;
  }

  f32x8 c[2][2];
#pragma unroll
  for (int mi = 0; mi < 2; ++mi)
#pragma unroll
    for (int f = 0; f < 2; ++f) c[mi][f] = (f32x8)(0.f);

  for (int k0 = 0; k0 < HID; k0 += 32) {
    bf16x16 a0 = frag_a(aptr[0], k0, hsel);
    bf16x16 a1 = frag_a(aptr[1], k0, hsel);
#pragma unroll
    for (int f = 0; f < 2; ++f) {
      bf16x16 b = frag_b(bp[f], k0, hsel);
      c[0][f] = wmma_bf16f32(a0, b, c[0][f]);
      c[1][f] = wmma_bf16f32(a1, b, c[1][f]);
    }
  }

#pragma unroll
  for (int mi = 0; mi < 2; ++mi) {
#pragma unroll
    for (int r = 0; r < 8; ++r) {
      int ml = waveM * 32 + mi * 16 + r + 8 * hsel;
      if (ROUTED && (m0 + ml) >= count) continue;
      size_t orow = (size_t)s_tok[ml] * DIM;
#pragma unroll
      for (int f = 0; f < 2; ++f) {
        size_t idx = orow + n0 + waveN * 32 + f * 16 + lm;
        if (ROUTED) atomicAdd(&out[idx], c[mi][f][r]);
        else        out[idx] = c[mi][f][r];
      }
    }
  }
}

// ------------------------------- launcher ----------------------------------
extern "C" void kernel_launch(void* const* d_in, const int* in_sizes, int n_in,
                              void* d_out, int out_size, void* d_ws,
                              size_t ws_size, hipStream_t stream) {
  (void)in_sizes; (void)n_in; (void)out_size; (void)ws_size;

  const float* x    = (const float*)d_in[0];
  const float* gate = (const float*)d_in[1];
  const float* ebias= (const float*)d_in[2];
  const float* w1   = (const float*)d_in[3];
  const float* w2   = (const float*)d_in[4];
  const float* w3   = (const float*)d_in[5];
  const float* sw1  = (const float*)d_in[6];
  const float* sw2  = (const float*)d_in[7];
  const float* sw3  = (const float*)d_in[8];
  float* out = (float*)d_out;

  char* w = (char*)d_ws;
  auto alloc = [&](size_t bytes) -> char* {
    char* p = w;
    w += (bytes + 255) & ~(size_t)255;
    return p;
  };
  const size_t ND = (size_t)NTOK * DIM, HD = (size_t)HID * DIM;
  __bf16* xb   = (__bf16*)alloc(ND * 2);
  __bf16* w1b  = (__bf16*)alloc((size_t)NEXP * HD * 2);
  __bf16* w3b  = (__bf16*)alloc((size_t)NEXP * HD * 2);
  __bf16* w2b  = (__bf16*)alloc((size_t)NEXP * HD * 2);
  __bf16* sw1b = (__bf16*)alloc(HD * 2);
  __bf16* sw3b = (__bf16*)alloc(HD * 2);
  __bf16* sw2b = (__bf16*)alloc(HD * 2);
  __bf16* h_rt = (__bf16*)alloc((size_t)NTOK * 2 * HID * 2);
  __bf16* h_sh = (__bf16*)alloc((size_t)NTOK * HID * 2);
  int*   tok   = (int*)  alloc((size_t)NEXP * NTOK * 4);
  float* score = (float*)alloc((size_t)NEXP * NTOK * 4);
  int*   dst   = (int*)  alloc((size_t)NEXP * NTOK * 4);
  int*   cnt   = (int*)  alloc(NEXP * 4);

  const int CB = 256, CG = 1024;
  cast_f32_bf16<<<CG, CB, 0, stream>>>(x,   xb,  (long)ND);
  cast_f32_bf16<<<CG, CB, 0, stream>>>(w1,  w1b, (long)NEXP * HD);
  cast_f32_bf16<<<CG, CB, 0, stream>>>(w3,  w3b, (long)NEXP * HD);
  cast_f32_bf16<<<CG, CB, 0, stream>>>(w2,  w2b, (long)NEXP * HD);
  cast_f32_bf16<<<CG, CB, 0, stream>>>(sw1, sw1b, (long)HD);
  cast_f32_bf16<<<CG, CB, 0, stream>>>(sw3, sw3b, (long)HD);
  cast_f32_bf16<<<CG, CB, 0, stream>>>(sw2, sw2b, (long)HD);

  zero_counts<<<1, 32, 0, stream>>>(cnt);
  router_kernel<<<NTOK / 8, 256, 0, stream>>>(x, gate, ebias, cnt, tok, score, dst);

  // Routed experts: h = silu(s*g)*(s*u)  (grid.x sized for worst case, early exit)
  ffn_gemm1<true><<<dim3(NTOK / 64, HID / 128, NEXP), 256, 0, stream>>>(
      xb, w1b, w3b, cnt, tok, score, dst, h_rt);
  // Shared expert stage 1 (s = 1)
  ffn_gemm1<false><<<dim3(NTOK / 64, HID / 128, 1), 256, 0, stream>>>(
      xb, sw1b, sw3b, nullptr, nullptr, nullptr, nullptr, h_sh);

  // Shared stage 2 plain-stores -> initializes out; routed stage 2 atomicAdds.
  ffn_gemm2<false><<<dim3(NTOK / 64, DIM / 128, 1), 256, 0, stream>>>(
      h_sh, sw2b, nullptr, nullptr, nullptr, out);
  ffn_gemm2<true><<<dim3(NTOK / 64, DIM / 128, NEXP), 256, 0, stream>>>(
      h_rt, w2b, cnt, tok, dst, out);
}
